// MoE_FiLM_Modulation_67405216743457
// MI455X (gfx1250) — matrix operand order repaired and checked by reading, not compiled, not run on hardware
//
#include <hip/hip_runtime.h>
#include <math.h>

// ---------------- problem constants ----------------
#define BATCH 8
#define SEQ   2048
#define DIM   1024
#define COND  512
#define NEXP  8
#define TOPK  4
#define MROWS (BATCH*SEQ)   // 16384

// ---------------- GEMM tiling ----------------
#define BM 128
#define BN 128
#define BK 32
#define LK 40   // padded LDS K-stride (ushorts): 80B rows, kills bank conflicts

typedef __attribute__((ext_vector_type(16))) __bf16 v16bf;
typedef __attribute__((ext_vector_type(8)))  __bf16 v8bf;
typedef __attribute__((ext_vector_type(8)))  float  v8f;

union Frag { v16bf v; v8bf h[2]; };

static __device__ __forceinline__ unsigned short f2bf(float f) {
    unsigned int u = __float_as_uint(f);
    u = (u + 0x7FFFu + ((u >> 16) & 1u)) >> 16;   // round-to-nearest-even
    return (unsigned short)u;
}
static __device__ __forceinline__ float bf2f(unsigned short h) {
    return __uint_as_float(((unsigned int)h) << 16);
}
static __device__ __forceinline__ float gelu_exact(float z) {
    return 0.5f * z * (1.0f + erff(z * 0.70710678118654752f));
}

// ============================================================
// Router: logits = cond@W + b; keep-top-4 (zero non-topk logits);
// softmax over all E (zeros included). One block per batch row.
// ============================================================
__global__ void router_kernel(const float* __restrict__ cond,
                              const float* __restrict__ rW,
                              const float* __restrict__ rb,
                              float* __restrict__ wout,
                              float* __restrict__ wtail) {
    __shared__ float lg[NEXP];
    const int b = blockIdx.x, lane = threadIdx.x;
    if (lane < NEXP) {
        float s = rb[lane];
        for (int c = 0; c < COND; ++c)
            s += cond[b * COND + c] * rW[c * NEXP + lane];
        lg[lane] = s;
    }
    __syncthreads();
    if (lane < NEXP) {
        float vals[NEXP];
        float mx = -1e30f;
        #pragma unroll
        for (int e = 0; e < NEXP; ++e) {
            float oe = lg[e];
            int r = 0;
            #pragma unroll
            for (int e2 = 0; e2 < NEXP; ++e2) {
                float o2 = lg[e2];
                r += (o2 > oe) || (o2 == oe && e2 < e);   // stable top-k rank
            }
            float mv = (r < TOPK) ? oe : 0.0f;            // logits * mask
            vals[e] = mv;
            mx = fmaxf(mx, mv);
        }
        float den = 0.0f;
        #pragma unroll
        for (int e = 0; e < NEXP; ++e) den += __expf(vals[e] - mx);
        float w = __expf(vals[lane] - mx) / den;
        wout[b * NEXP + lane]  = w;
        wtail[b * NEXP + lane] = w;
    }
}

// ============================================================
// Combined biases: cbg[b,n] = sum_e w[b,e]*bg[e,n] (and cbb)
// ============================================================
__global__ void combine_bias_kernel(const float* __restrict__ w,
                                    const float* __restrict__ bg,
                                    const float* __restrict__ bb,
                                    float* __restrict__ cbg,
                                    float* __restrict__ cbb) {
    int idx = blockIdx.x * blockDim.x + threadIdx.x;   // B*D
    int b = idx >> 10, n = idx & (DIM - 1);
    float sg = 0.f, sb = 0.f;
    #pragma unroll
    for (int e = 0; e < NEXP; ++e) {
        float we = w[b * NEXP + e];
        sg += we * bg[e * DIM + n];
        sb += we * bb[e * DIM + n];
    }
    cbg[idx] = sg;
    cbb[idx] = sb;
}

// ============================================================
// Combined expert weights, stored bf16 TRANSPOSED:
//   CWt[b][n][k] = sum_e w[b,e] * W[e][k][n]
// Thread handles (k0..k0+3, n); reads coalesced over n.
// ============================================================
__global__ __launch_bounds__(256)
void combine_experts_kernel(const float* __restrict__ w,
                            const float* __restrict__ Wg,
                            const float* __restrict__ Wb,
                            unsigned short* __restrict__ CWgT,
                            unsigned short* __restrict__ CWbT) {
    __shared__ float wsh[BATCH * NEXP];
    if (threadIdx.x < BATCH * NEXP) wsh[threadIdx.x] = w[threadIdx.x];
    __syncthreads();
    int idx = blockIdx.x * blockDim.x + threadIdx.x;   // D*(D/4)
    int n  = idx & (DIM - 1);
    int k0 = (idx >> 10) << 2;
    float gv[NEXP][4], bv[NEXP][4];
    #pragma unroll
    for (int e = 0; e < NEXP; ++e)
        #pragma unroll
        for (int i = 0; i < 4; ++i) {
            size_t off = (size_t)e * DIM * DIM + (size_t)(k0 + i) * DIM + n;
            gv[e][i] = Wg[off];
            bv[e][i] = Wb[off];
        }
    #pragma unroll
    for (int b = 0; b < BATCH; ++b) {
        float sg[4] = {0.f, 0.f, 0.f, 0.f}, sb[4] = {0.f, 0.f, 0.f, 0.f};
        #pragma unroll
        for (int e = 0; e < NEXP; ++e) {
            float we = wsh[b * NEXP + e];
            #pragma unroll
            for (int i = 0; i < 4; ++i) { sg[i] += we * gv[e][i]; sb[i] += we * bv[e][i]; }
        }
        size_t off = (size_t)b * DIM * DIM + (size_t)n * DIM + k0;
        ushort4 pg = make_ushort4(f2bf(sg[0]), f2bf(sg[1]), f2bf(sg[2]), f2bf(sg[3]));
        ushort4 pb = make_ushort4(f2bf(sb[0]), f2bf(sb[1]), f2bf(sb[2]), f2bf(sb[3]));
        *reinterpret_cast<ushort4*>(CWgT + off) = pg;
        *reinterpret_cast<ushort4*>(CWbT + off) = pb;
    }
}

// ============================================================
// fp32 [K][N] weight -> bf16 transposed [N][K]
// ============================================================
__global__ __launch_bounds__(256)
void convert_transpose_kernel(const float* __restrict__ W,
                              unsigned short* __restrict__ Wt) {
    int idx = blockIdx.x * blockDim.x + threadIdx.x;   // D*(D/4)
    int n  = idx & (DIM - 1);
    int k0 = (idx >> 10) << 2;
    float v[4];
    #pragma unroll
    for (int i = 0; i < 4; ++i) v[i] = W[(size_t)(k0 + i) * DIM + n];
    ushort4 p = make_ushort4(f2bf(v[0]), f2bf(v[1]), f2bf(v[2]), f2bf(v[3]));
    *reinterpret_cast<ushort4*>(Wt + (size_t)n * DIM + k0) = p;
}

// ============================================================
// LayerNorm over D=1024, one block per row, bf16 out
// ============================================================
__global__ __launch_bounds__(256)
void ln_kernel(const float* __restrict__ feat,
               const float* __restrict__ g,
               const float* __restrict__ bias,
               unsigned short* __restrict__ out) {
    __shared__ float s1[256], s2[256];
    const int row = blockIdx.x, tid = threadIdx.x;
    const int col = tid * 4;
    float4 v = *reinterpret_cast<const float4*>(feat + (size_t)row * DIM + col);
    float sum = v.x + v.y + v.z + v.w;
    float sq  = v.x * v.x + v.y * v.y + v.z * v.z + v.w * v.w;
    s1[tid] = sum; s2[tid] = sq;
    __syncthreads();
    for (int off = 128; off > 0; off >>= 1) {
        if (tid < off) { s1[tid] += s1[tid + off]; s2[tid] += s2[tid + off]; }
        __syncthreads();
    }
    const float mu  = s1[0] * (1.0f / DIM);
    const float var = s2[0] * (1.0f / DIM) - mu * mu;
    const float rs  = rsqrtf(var + 1e-5f);
    float4 gg = *reinterpret_cast<const float4*>(g + col);
    float4 bb = *reinterpret_cast<const float4*>(bias + col);
    unsigned short r[4];
    r[0] = f2bf((v.x - mu) * rs * gg.x + bb.x);
    r[1] = f2bf((v.y - mu) * rs * gg.y + bb.y);
    r[2] = f2bf((v.z - mu) * rs * gg.z + bb.z);
    r[3] = f2bf((v.w - mu) * rs * gg.w + bb.w);
    *reinterpret_cast<ushort4*>(out + (size_t)row * DIM + col) =
        make_ushort4(r[0], r[1], r[2], r[3]);
}

// ============================================================
// Generic bf16 WMMA GEMM, C = A[M,K] @ Bt[N,K]^T + bias, epilogue:
//   EPI=0: out_bf16 = gelu(C)           (base_net / adaptor-1)
//   EPI=1: out_f32  = C + resid         (adaptor-2 + residual)
// Block 256 (8 wave32 as 2x4), each wave: 4 Mx2 N 16x16 tiles.
// ============================================================
template <int EPI>
__global__ __launch_bounds__(256)
void gemm_bf16_epi(const unsigned short* __restrict__ A,
                   const unsigned short* __restrict__ Bt,
                   const float* __restrict__ bias,
                   unsigned short* __restrict__ outBf,
                   float* __restrict__ outF,
                   const float* __restrict__ resid,
                   int K, int N) {
    __shared__ unsigned short As[BM * LK];
    __shared__ unsigned short Bs[BN * LK];
    const int tid = threadIdx.x;
    const int m0 = blockIdx.x * BM;
    const int n0 = blockIdx.y * BN;
    const int lane = tid & 31;
    const int w = tid >> 5;
    const int wm = w & 1, wn = w >> 1;
    const int l15 = lane & 15, hi = lane >> 4;

    v8f acc[4][2];
    #pragma unroll
    for (int i = 0; i < 4; ++i)
        #pragma unroll
        for (int j = 0; j < 2; ++j)
            #pragma unroll
            for (int v = 0; v < 8; ++v) acc[i][j][v] = 0.0f;

    for (int k0 = 0; k0 < K; k0 += BK) {
        #pragma unroll
        for (int t = 0; t < 2; ++t) {
            int c = tid + t * 256;
            int row = c >> 2;
            int col = (c & 3) << 3;
            *reinterpret_cast<float4*>(&As[row * LK + col]) =
                *reinterpret_cast<const float4*>(A + (size_t)(m0 + row) * K + k0 + col);
            *reinterpret_cast<float4*>(&Bs[row * LK + col]) =
                *reinterpret_cast<const float4*>(Bt + (size_t)(n0 + row) * K + k0 + col);
        }
        __syncthreads();
        Frag fa[4], fb[2];
        #pragma unroll
        for (int i = 0; i < 4; ++i) {
            int r = wm * 64 + i * 16 + l15;
            int kb = hi * 16;   // A k order used here (B matches, not ISA table): lane<16 -> K 0..15, lane>=16 -> K 16..31
            fa[i].h[0] = *reinterpret_cast<const v8bf*>(&As[r * LK + kb]);
            fa[i].h[1] = *reinterpret_cast<const v8bf*>(&As[r * LK + kb + 8]);
        }
        #pragma unroll
        for (int j = 0; j < 2; ++j) {
            int r = wn * 32 + j * 16 + l15;
            int kb = hi * 16;   // B k order, same as A: lane<16 -> K 0..15, lane>=16 -> K 16..31
            fb[j].h[0] = *reinterpret_cast<const v8bf*>(&Bs[r * LK + kb]);
            fb[j].h[1] = *reinterpret_cast<const v8bf*>(&Bs[r * LK + kb + 8]);
        }
        #pragma unroll
        for (int i = 0; i < 4; ++i)
            #pragma unroll
            for (int j = 0; j < 2; ++j)
                acc[i][j] = __builtin_amdgcn_wmma_f32_16x16x32_bf16(
                    false, fa[i].v, false, fb[j].v, (short)0, acc[i][j], false, false);
        __syncthreads();
    }

    // Epilogue: C/D layout: lane n = l15, m = vgpr + (lane>=16 ? 8 : 0)
    #pragma unroll
    for (int i = 0; i < 4; ++i)
        #pragma unroll
        for (int j = 0; j < 2; ++j) {
            const int gm0 = m0 + wm * 64 + i * 16;
            const int gn  = n0 + wn * 32 + j * 16 + l15;
            const float bv = bias[gn];
            #pragma unroll
            for (int v = 0; v < 8; ++v) {
                const int m = gm0 + v + hi * 8;
                const float val = acc[i][j][v] + bv;
                if constexpr (EPI == 0) {
                    outBf[(size_t)m * N + gn] = f2bf(gelu_exact(val));
                } else {
                    outF[(size_t)m * N + gn] = val + resid[(size_t)m * N + gn];
                }
            }
        }
}

// ============================================================
// FiLM dual-GEMM per batch b:
//   gamma = x_b @ CWg_b^T + cbg_b ; beta = x_b @ CWb_b^T + cbb_b
//   y = gamma * x + beta   (bf16 out)
// Shares A(x) tiles; 16 WMMAs per k-step per wave.
// ============================================================
__global__ __launch_bounds__(256)
void film_gemm(const unsigned short* __restrict__ Xbf,
               const unsigned short* __restrict__ CWgT,
               const unsigned short* __restrict__ CWbT,
               const float* __restrict__ cbg,
               const float* __restrict__ cbb,
               unsigned short* __restrict__ Ybf) {
    __shared__ unsigned short As[BM * LK];
    __shared__ unsigned short Gs[BN * LK];
    __shared__ unsigned short Hs[BN * LK];
    const int b = blockIdx.z;
    const unsigned short* A  = Xbf  + (size_t)b * SEQ * DIM;
    const unsigned short* Bg = CWgT + (size_t)b * DIM * DIM;
    const unsigned short* Bb = CWbT + (size_t)b * DIM * DIM;
    const float* bg = cbg + (size_t)b * DIM;
    const float* bbv = cbb + (size_t)b * DIM;
    unsigned short* Y = Ybf + (size_t)b * SEQ * DIM;

    const int tid = threadIdx.x;
    const int m0 = blockIdx.x * BM;
    const int n0 = blockIdx.y * BN;
    const int lane = tid & 31;
    const int w = tid >> 5;
    const int wm = w & 1, wn = w >> 1;
    const int l15 = lane & 15, hi = lane >> 4;

    v8f accg[4][2], accb[4][2];
    #pragma unroll
    for (int i = 0; i < 4; ++i)
        #pragma unroll
        for (int j = 0; j < 2; ++j)
            #pragma unroll
            for (int v = 0; v < 8; ++v) { accg[i][j][v] = 0.0f; accb[i][j][v] = 0.0f; }

    for (int k0 = 0; k0 < DIM; k0 += BK) {
        #pragma unroll
        for (int t = 0; t < 2; ++t) {
            int c = tid + t * 256;
            int row = c >> 2;
            int col = (c & 3) << 3;
            *reinterpret_cast<float4*>(&As[row * LK + col]) =
                *reinterpret_cast<const float4*>(A + (size_t)(m0 + row) * DIM + k0 + col);
            *reinterpret_cast<float4*>(&Gs[row * LK + col]) =
                *reinterpret_cast<const float4*>(Bg + (size_t)(n0 + row) * DIM + k0 + col);
            *reinterpret_cast<float4*>(&Hs[row * LK + col]) =
                *reinterpret_cast<const float4*>(Bb + (size_t)(n0 + row) * DIM + k0 + col);
        }
        __syncthreads();
        Frag fa[4], fg[2], fh[2];
        #pragma unroll
        for (int i = 0; i < 4; ++i) {
            int r = wm * 64 + i * 16 + l15;
            int kb = hi * 16;
            fa[i].h[0] = *reinterpret_cast<const v8bf*>(&As[r * LK + kb]);
            fa[i].h[1] = *reinterpret_cast<const v8bf*>(&As[r * LK + kb + 8]);
        }
        #pragma unroll
        for (int j = 0; j < 2; ++j) {
            int r = wn * 32 + j * 16 + l15;
            int kb = hi * 16;
            fg[j].h[0] = *reinterpret_cast<const v8bf*>(&Gs[r * LK + kb]);
            fg[j].h[1] = *reinterpret_cast<const v8bf*>(&Gs[r * LK + kb + 8]);
            fh[j].h[0] = *reinterpret_cast<const v8bf*>(&Hs[r * LK + kb]);
            fh[j].h[1] = *reinterpret_cast<const v8bf*>(&Hs[r * LK + kb + 8]);
        }
        #pragma unroll
        for (int i = 0; i < 4; ++i)
            #pragma unroll
            for (int j = 0; j < 2; ++j) {
                accg[i][j] = __builtin_amdgcn_wmma_f32_16x16x32_bf16(
                    false, fa[i].v, false, fg[j].v, (short)0, accg[i][j], false, false);
                accb[i][j] = __builtin_amdgcn_wmma_f32_16x16x32_bf16(
                    false, fa[i].v, false, fh[j].v, (short)0, accb[i][j], false, false);
            }
        __syncthreads();
    }

    #pragma unroll
    for (int i = 0; i < 4; ++i)
        #pragma unroll
        for (int j = 0; j < 2; ++j) {
            const int gm0 = m0 + wm * 64 + i * 16;
            const int gn  = n0 + wn * 32 + j * 16 + l15;
            const float gbias = bg[gn];
            const float bbias = bbv[gn];
            #pragma unroll
            for (int v = 0; v < 8; ++v) {
                const int m = gm0 + v + hi * 8;
                const float xf = bf2f(A[(size_t)m * DIM + gn]);
                const float y = (accg[i][j][v] + gbias) * xf + (accb[i][j][v] + bbias);
                Y[(size_t)m * DIM + gn] = f2bf(y);
            }
        }
}

// ============================================================
// Launcher
// ============================================================
extern "C" void kernel_launch(void* const* d_in, const int* in_sizes, int n_in,
                              void* d_out, int out_size, void* d_ws, size_t ws_size,
                              hipStream_t stream) {
    const float* features = (const float*)d_in[0];
    const float* condition= (const float*)d_in[1];
    const float* router_W = (const float*)d_in[2];
    const float* router_b = (const float*)d_in[3];
    const float* Wg       = (const float*)d_in[4];
    const float* bg       = (const float*)d_in[5];
    const float* Wb       = (const float*)d_in[6];
    const float* bb       = (const float*)d_in[7];
    const float* ln_g     = (const float*)d_in[8];
    const float* ln_b     = (const float*)d_in[9];
    const float* base_W   = (const float*)d_in[10];
    const float* base_b   = (const float*)d_in[11];
    const float* ad_W1    = (const float*)d_in[12];
    const float* ad_b1    = (const float*)d_in[13];
    const float* ad_W2    = (const float*)d_in[14];
    const float* ad_b2    = (const float*)d_in[15];
    float* out = (float*)d_out;

    // ---- workspace layout (bytes) ----
    char* ws = (char*)d_ws;
    const size_t SZ_ACT_BF = (size_t)MROWS * DIM * 2;           // 32 MB
    const size_t SZ_CW_BF  = (size_t)BATCH * DIM * DIM * 2;     // 16 MB
    const size_t SZ_W_BF   = (size_t)DIM * DIM * 2;             // 2 MB
    size_t o = 0;
    unsigned short* lnh_buf = (unsigned short*)(ws + o); o += SZ_ACT_BF;  // ln_x, reused as h
    unsigned short* x_bf    = (unsigned short*)(ws + o); o += SZ_ACT_BF;
    unsigned short* y_bf    = (unsigned short*)(ws + o); o += SZ_ACT_BF;
    unsigned short* cwg_t   = (unsigned short*)(ws + o); o += SZ_CW_BF;
    unsigned short* cwb_t   = (unsigned short*)(ws + o); o += SZ_CW_BF;
    unsigned short* bw_t    = (unsigned short*)(ws + o); o += SZ_W_BF;
    unsigned short* a1_t    = (unsigned short*)(ws + o); o += SZ_W_BF;
    unsigned short* a2_t    = (unsigned short*)(ws + o); o += SZ_W_BF;
    float* cbg   = (float*)(ws + o); o += (size_t)BATCH * DIM * 4;
    float* cbb   = (float*)(ws + o); o += (size_t)BATCH * DIM * 4;
    float* wvec  = (float*)(ws + o); o += 256;

    float* weights_out = out + (size_t)MROWS * DIM;   // tuple output #2

    // 1) router
    router_kernel<<<BATCH, 32, 0, stream>>>(condition, router_W, router_b, wvec, weights_out);
    // 2) combined biases
    combine_bias_kernel<<<(BATCH * DIM) / 256, 256, 0, stream>>>(wvec, bg, bb, cbg, cbb);
    // 3) combined expert weights (bf16 transposed)
    combine_experts_kernel<<<(DIM * (DIM / 4)) / 256, 256, 0, stream>>>(wvec, Wg, Wb, cwg_t, cwb_t);
    // 4) static weight converts (bf16 transposed)
    convert_transpose_kernel<<<(DIM * (DIM / 4)) / 256, 256, 0, stream>>>(base_W, bw_t);
    convert_transpose_kernel<<<(DIM * (DIM / 4)) / 256, 256, 0, stream>>>(ad_W1, a1_t);
    convert_transpose_kernel<<<(DIM * (DIM / 4)) / 256, 256, 0, stream>>>(ad_W2, a2_t);
    // 5) LayerNorm -> bf16
    ln_kernel<<<MROWS, 256, 0, stream>>>(features, ln_g, ln_b, lnh_buf);
    // 6) base_net GEMM: x = gelu(ln_x @ base_W + b)
    gemm_bf16_epi<0><<<dim3(MROWS / BM, DIM / BN), 256, 0, stream>>>(
        lnh_buf, bw_t, base_b, x_bf, nullptr, nullptr, DIM, DIM);
    // 7) FiLM dual GEMM per batch: y
    film_gemm<<<dim3(SEQ / BM, DIM / BN, BATCH), 256, 0, stream>>>(
        x_bf, cwg_t, cwb_t, cbg, cbb, y_bf);
    // 8) adaptor GEMM1: h = gelu(y @ ad_W1 + b1)   (h reuses ln buffer)
    gemm_bf16_epi<0><<<dim3(MROWS / BM, DIM / BN), 256, 0, stream>>>(
        y_bf, a1_t, ad_b1, lnh_buf, nullptr, nullptr, DIM, DIM);
    // 9) adaptor GEMM2 + residual: out = h @ ad_W2 + b2 + features
    gemm_bf16_epi<1><<<dim3(MROWS / BM, DIM / BN), 256, 0, stream>>>(
        lnh_buf, a2_t, ad_b2, nullptr, out, features, DIM, DIM);
    (void)in_sizes; (void)n_in; (void)out_size; (void)ws_size;
}
